// ConvModule_30975304139353
// MI455X (gfx1250) — compile-verified, matrix operand order checked by reading, not run on hardware
//
#include <hip/hip_runtime.h>
#include <hip/hip_bf16.h>

// ---------------------------------------------------------------------------
// Conformer conv module on gfx1250 (MI455X), compile-only target.
// GEMMs run on v_wmma_f32_16x16x32_bf16 (f32 accumulate). Intermediates are
// bf16 in d_ws (2 x 32MB) and live in the 192MB L2. Weights are pre-packed
// into WMMA fragment lane order so each lane loads 32 contiguous bytes.
// Each wave covers all 4 M-tiles of its block so every B fragment is reused
// by 4 WMMAs -> each block streams the packed weights exactly once.
// The p (n-tile) loops are kept rolled with a memory-clobber so the compiler
// cannot hoist p-invariant LDS A-fragments (which caused scratch spills).
// ---------------------------------------------------------------------------

typedef __attribute__((ext_vector_type(16))) __bf16 bf16x16;
typedef __attribute__((ext_vector_type(8)))  float  v8f;
typedef __attribute__((ext_vector_type(4)))  float  f32x4;
typedef __attribute__((ext_vector_type(4)))  unsigned int u32x4;

union Frag   { bf16x16 v; u32x4 q[2]; };
union Pack4  { u32x4 q; unsigned int u[4]; };
union Pack16 { u32x4 q[2]; unsigned short h[16]; };

__device__ __forceinline__ unsigned short f2bf(float f) {
  unsigned int u = __float_as_uint(f);
  unsigned int r = (u + 0x7FFFu + ((u >> 16) & 1u)) >> 16;   // RNE
  return (unsigned short)r;
}
__device__ __forceinline__ float bf2f(unsigned short h) {
  return __uint_as_float(((unsigned int)h) << 16);
}

// ---------------- workspace layout (bytes) ----------------
#define W1P_OFF   ((size_t)0)                       // 512*1024 bf16 = 1 MB
#define W2P_OFF   ((size_t)1048576)                 // 512*512  bf16 = 512 KB
#define DWK_OFF   ((size_t)1572864)                 // 32*512   bf16 = 32 KB
#define BNS_OFF   ((size_t)1605632)                 // 512 f32
#define BNSH_OFF  ((size_t)1607680)                 // 512 f32
#define HMID_OFF  ((size_t)1609728)                 // 16*2048*512 bf16 = 32 MB
#define ACT_OFF   ((size_t)35164160)                // 16*2048*512 bf16 = 32 MB

// ---------------------------------------------------------------------------
// Pack weights [K=512][N] f32 row-major -> per-fragment lane-ordered bf16.
// Fragment (kb, nb) covers K kb*32..+31, N nb*16..+15. Lane l<16: col nb*16+l,
// K {0..7,16..23}; lane>=16: col nb*16+l-16, K {8..15,24..31}. 16 bf16/lane.
// ---------------------------------------------------------------------------
__global__ void pack_w_kernel(const float* __restrict__ src,
                              unsigned short* __restrict__ dst,
                              int N, int ntiles) {
  int gid = blockIdx.x * blockDim.x + threadIdx.x;
  int total = 16 * ntiles * 32;                 // (512/32) k-tiles
  if (gid >= total) return;
  int lane = gid & 31;
  int frag = gid >> 5;
  int kb = frag / ntiles, nb = frag % ntiles;
  int n  = nb * 16 + (lane & 15);
  int k0 = kb * 32 + ((lane < 16) ? 0 : 8);
  unsigned short* o = dst + ((size_t)frag << 9) + (lane << 4);
#pragma unroll
  for (int e = 0; e < 16; ++e) {
    int k = k0 + e + ((e < 8) ? 0 : 8);
    o[e] = f2bf(src[(size_t)k * N + n]);
  }
}

__global__ void pack_misc_kernel(const float* __restrict__ dwk,
                                 const float* __restrict__ bg,
                                 const float* __restrict__ bb,
                                 const float* __restrict__ bm,
                                 const float* __restrict__ bv,
                                 unsigned short* __restrict__ dwkb,
                                 float* __restrict__ bns,
                                 float* __restrict__ bnsh) {
  int gid = blockIdx.x * blockDim.x + threadIdx.x;
  if (gid < 32 * 512) dwkb[gid] = f2bf(dwk[gid]);
  if (gid < 512) {
    float s = bg[gid] * rsqrtf(bv[gid] + 1e-3f);
    bns[gid]  = s;
    bnsh[gid] = bb[gid] - bm[gid] * s;
  }
}

// ---------------------------------------------------------------------------
// Kernel 1: LayerNorm + GEMM1 (512->1024, bf16 WMMA) + GLU -> hmid bf16.
// Block = 256 thr (8 waves), 64 rows. Wave w handles GLU pair-tiles
// w*4..w*4+3 across all 4 M-tiles: each (bA,bG) fragment pair feeds 8 WMMAs.
// GLU pairs (n, n+512) are computed together so gating stays in registers.
// ---------------------------------------------------------------------------
__global__ __launch_bounds__(256) void k1_ln_gemm_glu(
    const float* __restrict__ x, const float* __restrict__ gam,
    const float* __restrict__ bet, const unsigned short* __restrict__ w1p,
    const float* __restrict__ b1, unsigned short* __restrict__ hmid) {
  __shared__ unsigned short lds_h[64 * 512];   // 64 KB, swizzled bf16 tile
  const int tid  = threadIdx.x;
  const int row0 = blockIdx.x * 64;

  // ---- LayerNorm: 4 threads per row, 128 contiguous elems each ----
  const int rl = tid >> 2, part = tid & 3;
  const float* xr = x + (size_t)(row0 + rl) * 512 + part * 128;
  float s = 0.f, s2 = 0.f;
#pragma unroll
  for (int i = 0; i < 128; i += 4) {
    f32x4 v = *(const f32x4*)(xr + i);
    s  += v.x + v.y + v.z + v.w;
    s2 += v.x * v.x + v.y * v.y + v.z * v.z + v.w * v.w;
  }
  s  += __shfl_xor(s, 1);  s  += __shfl_xor(s, 2);
  s2 += __shfl_xor(s2, 1); s2 += __shfl_xor(s2, 2);
  const float mean = s * (1.f / 512.f);
  const float var  = s2 * (1.f / 512.f) - mean * mean;
  const float rstd = rsqrtf(var + 1e-3f);
#pragma unroll
  for (int i = 0; i < 128; i += 8) {
    int k = part * 128 + i;
    Pack4 pk;
#pragma unroll
    for (int e = 0; e < 8; e += 2) {
      float h0 = (xr[i + e]     - mean) * rstd * gam[k + e]     + bet[k + e];
      float h1 = (xr[i + e + 1] - mean) * rstd * gam[k + e + 1] + bet[k + e + 1];
      pk.u[e >> 1] = (unsigned)f2bf(h0) | ((unsigned)f2bf(h1) << 16);
    }
    int ch = k >> 3;                              // 16B chunk index 0..63
    *(u32x4*)&lds_h[(rl << 9) + ((ch ^ (rl & 7)) << 3)] = pk.q;
  }
  __syncthreads();

  // ---- WMMA GEMM + GLU ----
  const int wave = tid >> 5, lane = tid & 31;
  const int arl  = lane & 15;                     // A row within M-tile
  const int sw   = lane & 7;                      // chunk swizzle (m-invariant)
  const int asel = (lane < 16) ? 0 : 1;           // A k-chunk select
  const int colL = lane & 15;

#pragma unroll 1
  for (int p = 0; p < 4; ++p) {
    asm volatile("" ::: "memory");                // block LDS-load hoisting
    const int pt = (wave << 2) + p;               // GLU pair-tile 0..31
    v8f accA[4] = {};
    v8f accG[4] = {};
#pragma unroll 1
    for (int kt = 0; kt < 16; ++kt) {
      int c0 = (kt << 2) + asel;
      int o0 = ((c0      ) ^ sw) << 3;
      int o1 = ((c0 + 2) ^ sw) << 3;
      Frag a[4];
#pragma unroll
      for (int m = 0; m < 4; ++m) {
        const unsigned short* ab = &lds_h[((m << 4) + arl) << 9];
        a[m].q[0] = *(const u32x4*)&ab[o0];
        a[m].q[1] = *(const u32x4*)&ab[o1];
      }
      const unsigned short* bA = w1p + (((size_t)(kt * 64 + pt))      << 9) + (lane << 4);
      const unsigned short* bG = w1p + (((size_t)(kt * 64 + pt + 32)) << 9) + (lane << 4);
      Frag fa, fg;
      fa.q[0] = *(const u32x4*)bA; fa.q[1] = *(const u32x4*)(bA + 8);
      fg.q[0] = *(const u32x4*)bG; fg.q[1] = *(const u32x4*)(bG + 8);
#pragma unroll
      for (int m = 0; m < 4; ++m) {
        accA[m] = __builtin_amdgcn_wmma_f32_16x16x32_bf16(
            false, a[m].v, false, fa.v, (short)0, accA[m], false, false);
        accG[m] = __builtin_amdgcn_wmma_f32_16x16x32_bf16(
            false, a[m].v, false, fg.v, (short)0, accG[m], false, false);
      }
    }
    // GLU epilogue: a * sigmoid(g) in registers, store bf16
    const float biasA = b1[(pt << 4) + colL];
    const float biasG = b1[512 + (pt << 4) + colL];
#pragma unroll
    for (int m = 0; m < 4; ++m) {
      int rbase = row0 + (m << 4) + ((lane >> 4) << 3);
#pragma unroll
      for (int v = 0; v < 8; ++v) {
        float aV  = accA[m][v] + biasA;
        float gV  = accG[m][v] + biasG;
        float glu = aV * (1.0f / (1.0f + __expf(-gV)));
        hmid[(size_t)(rbase + v) * 512 + (pt << 4) + colL] = f2bf(glu);
      }
    }
  }
}

// ---------------------------------------------------------------------------
// Kernel 2: depthwise conv (K=32, pad 15/16) + BN + swish. Block covers
// 128 t x 128 d of one batch; input halo staged in LDS (bf16), weights in LDS
// (f32). Each thread: 4 t x 16 ch with a fully-unrolled sliding row window so
// each input row is converted to f32 once and reused by 4 outputs.
// ---------------------------------------------------------------------------
__global__ __launch_bounds__(256) void k2_dwconv_bn_swish(
    const unsigned short* __restrict__ hmid,
    const unsigned short* __restrict__ dwkb,
    const float* __restrict__ dwb, const float* __restrict__ bns,
    const float* __restrict__ bnsh, unsigned short* __restrict__ act) {
  __shared__ unsigned short in_t[159 * 136];   // 128+31 rows, 128 ch + 8 pad
  __shared__ float wk_t[32 * 128];
  const int tid = threadIdx.x;
  const int t0 = blockIdx.x * 128;
  const int d0 = blockIdx.y * 128;
  const size_t seq = (size_t)blockIdx.z * 2048 * 512;

  for (int c = tid; c < 159 * 16; c += 256) {   // 16 x 16B per row
    int rr = c >> 4, off = c & 15;
    int gt = t0 - 15 + rr;
    u32x4 v = {0u, 0u, 0u, 0u};
    if (gt >= 0 && gt < 2048)
      v = *(const u32x4*)&hmid[seq + (size_t)gt * 512 + d0 + (off << 3)];
    *(u32x4*)&in_t[rr * 136 + (off << 3)] = v;
  }
  for (int c = tid; c < 32 * 128; c += 256)
    wk_t[c] = bf2f(dwkb[(c >> 7) * 512 + d0 + (c & 127)]);
  __syncthreads();

  const int dg = tid & 7, tg = tid >> 3;        // 16 ch, 4 t per thread
  const int chb = dg << 4;

  float acc[4][16] = {};
  float win[4][16];
  // helper: load + convert one staged row (16 ch)
  auto load_row = [&](int lr, float* w) {
    const unsigned short* ip = &in_t[lr * 136 + chb];
    Pack4 a, b;
    a.q = *(const u32x4*)ip;
    b.q = *(const u32x4*)(ip + 8);
#pragma unroll
    for (int e = 0; e < 4; ++e) {
      w[2 * e]     = __uint_as_float(a.u[e] << 16);
      w[2 * e + 1] = __uint_as_float(a.u[e] & 0xffff0000u);
      w[8 + 2 * e]     = __uint_as_float(b.u[e] << 16);
      w[8 + 2 * e + 1] = __uint_as_float(b.u[e] & 0xffff0000u);
    }
  };
  load_row((tg << 2) + 0, win[0]);
  load_row((tg << 2) + 1, win[1]);
  load_row((tg << 2) + 2, win[2]);
#pragma unroll
  for (int k = 0; k < 32; ++k) {
    load_row((tg << 2) + k + 3, win[(k + 3) & 3]);
    const float* wp = &wk_t[(k << 7) + chb];
    float wr[16];
#pragma unroll
    for (int c = 0; c < 16; ++c) wr[c] = wp[c];
#pragma unroll
    for (int o = 0; o < 4; ++o)
#pragma unroll
      for (int c = 0; c < 16; ++c)
        acc[o][c] += win[(k + o) & 3][c] * wr[c];
  }
#pragma unroll
  for (int o = 0; o < 4; ++o) {
    int t = t0 + (tg << 2) + o;
    Pack16 ob;
#pragma unroll
    for (int c = 0; c < 16; ++c) {
      int d = d0 + chb + c;
      float y = acc[o][c] + dwb[d];
      y = y * bns[d] + bnsh[d];
      y = y * (1.0f / (1.0f + __expf(-y)));     // swish
      ob.h[c] = f2bf(y);
    }
    u32x4* dst = (u32x4*)&act[seq + (size_t)t * 512 + d0 + chb];
    dst[0] = ob.q[0];
    dst[1] = ob.q[1];
  }
}

// ---------------------------------------------------------------------------
// Kernel 3: GEMM2 (512->512, bf16 WMMA) + bias + residual -> f32 out.
// Wave w handles n-tiles w*4..w*4+3 across all 4 M-tiles (B reused x4).
// ---------------------------------------------------------------------------
__global__ __launch_bounds__(256) void k3_gemm_res(
    const unsigned short* __restrict__ act,
    const unsigned short* __restrict__ w2p, const float* __restrict__ b2,
    const float* __restrict__ x, float* __restrict__ out) {
  __shared__ unsigned short lds_a[64 * 512];
  const int tid = threadIdx.x;
  const int row0 = blockIdx.x * 64;
  for (int c = tid; c < 4096; c += 256) {       // 64 rows x 64 chunks of 16B
    int rl = c >> 6, off = c & 63;
    u32x4 v = *(const u32x4*)&act[(size_t)(row0 + rl) * 512 + (off << 3)];
    *(u32x4*)&lds_a[(rl << 9) + ((off ^ (rl & 7)) << 3)] = v;
  }
  __syncthreads();

  const int wave = tid >> 5, lane = tid & 31;
  const int arl  = lane & 15;
  const int sw   = lane & 7;
  const int asel = (lane < 16) ? 0 : 1;
  const int colL = lane & 15;

#pragma unroll 1
  for (int p = 0; p < 4; ++p) {
    asm volatile("" ::: "memory");                // block LDS-load hoisting
    const int nt = (wave << 2) + p;               // n-tile 0..31
    v8f acc[4] = {};
#pragma unroll 1
    for (int kt = 0; kt < 16; ++kt) {
      int c0 = (kt << 2) + asel;
      int o0 = ((c0      ) ^ sw) << 3;
      int o1 = ((c0 + 2) ^ sw) << 3;
      Frag a[4];
#pragma unroll
      for (int m = 0; m < 4; ++m) {
        const unsigned short* ab = &lds_a[((m << 4) + arl) << 9];
        a[m].q[0] = *(const u32x4*)&ab[o0];
        a[m].q[1] = *(const u32x4*)&ab[o1];
      }
      const unsigned short* bp = w2p + (((size_t)(kt * 32 + nt)) << 9) + (lane << 4);
      Frag fb;
      fb.q[0] = *(const u32x4*)bp;
      fb.q[1] = *(const u32x4*)(bp + 8);
#pragma unroll
      for (int m = 0; m < 4; ++m)
        acc[m] = __builtin_amdgcn_wmma_f32_16x16x32_bf16(
            false, a[m].v, false, fb.v, (short)0, acc[m], false, false);
    }
    const float bias = b2[(nt << 4) + colL];
#pragma unroll
    for (int m = 0; m < 4; ++m) {
      int rbase = row0 + (m << 4) + ((lane >> 4) << 3);
#pragma unroll
      for (int v = 0; v < 8; ++v) {
        size_t idx = (size_t)(rbase + v) * 512 + (nt << 4) + colL;
        out[idx] = acc[m][v] + bias + x[idx];
      }
    }
  }
}

// ---------------------------------------------------------------------------
extern "C" void kernel_launch(void* const* d_in, const int* in_sizes, int n_in,
                              void* d_out, int out_size, void* d_ws,
                              size_t ws_size, hipStream_t stream) {
  const float* x        = (const float*)d_in[0];
  const float* ln_gamma = (const float*)d_in[1];
  const float* ln_beta  = (const float*)d_in[2];
  const float* w1       = (const float*)d_in[3];
  const float* b1       = (const float*)d_in[4];
  const float* dw_k     = (const float*)d_in[5];
  const float* dw_b     = (const float*)d_in[6];
  const float* bn_g     = (const float*)d_in[7];
  const float* bn_b     = (const float*)d_in[8];
  const float* bn_m     = (const float*)d_in[9];
  const float* bn_v     = (const float*)d_in[10];
  const float* w2       = (const float*)d_in[11];
  const float* b2       = (const float*)d_in[12];
  float* out = (float*)d_out;

  char* ws = (char*)d_ws;
  unsigned short* w1p  = (unsigned short*)(ws + W1P_OFF);
  unsigned short* w2p  = (unsigned short*)(ws + W2P_OFF);
  unsigned short* dwkb = (unsigned short*)(ws + DWK_OFF);
  float* bns  = (float*)(ws + BNS_OFF);
  float* bnsh = (float*)(ws + BNSH_OFF);
  unsigned short* hmid = (unsigned short*)(ws + HMID_OFF);
  unsigned short* actb = (unsigned short*)(ws + ACT_OFF);

  pack_w_kernel<<<128, 256, 0, stream>>>(w1, w1p, 1024, 64);
  pack_w_kernel<<<64, 256, 0, stream>>>(w2, w2p, 512, 32);
  pack_misc_kernel<<<64, 256, 0, stream>>>(dw_k, bn_g, bn_b, bn_m, bn_v,
                                           dwkb, bns, bnsh);
  k1_ln_gemm_glu<<<512, 256, 0, stream>>>(x, ln_gamma, ln_beta, w1p, b1, hmid);
  k2_dwconv_bn_swish<<<dim3(16, 4, 16), 256, 0, stream>>>(hmid, dwkb, dw_b,
                                                          bns, bnsh, actb);
  k3_gemm_res<<<512, 256, 0, stream>>>(actb, w2p, b2, x, out);
}
